// Quixer_39530878993000
// MI455X (gfx1250) — matrix operand order repaired and checked by reading, not compiled, not run on hardware
//
#include <hip/hip_runtime.h>
#include <hip/hip_bf16.h>

// ---------------------------------------------------------------------------
// Quixer forward on gfx1250 (MI455X).
//  - angles GEMM (1024x512 @ 512x80)  -> v_wmma_f32_16x16x32_f16
//  - quantum state sim: 1 workgroup (512 thr) per state, state in LDS (8KB)
//  - logits GEMM (32x512 @ 512x32000) -> v_wmma_f32_16x16x32_f16, streams w2
//    once (65.5MB -> HBM roofline ~2.8us @ 23.3 TB/s)
// ---------------------------------------------------------------------------

typedef __attribute__((ext_vector_type(16))) _Float16 v16h;
typedef __attribute__((ext_vector_type(8)))  float    v8f;

#define DIMQ   1024
#define NPAR   80
#define NTOK   32
#define NBATCH 32
#define VOCAB_ 32000
#define EMBD   512

// ---- workspace offsets (bytes) --------------------------------------------
// phase 1 (sim): angles | mono | acc | st
// phase 2 (post-sim): normv/ev/h16 overlay the dead angles region,
//                     normed overlays the dead mono region.
#define OFF_ANGLES 0u          // 1024*80*4   = 327680
#define OFF_MONO   327680u     // 32*1024*8   = 262144
#define OFF_ACC    589824u     // 262144
#define OFF_ST     851968u     // 1024*1024*8 = 8388608  (end 9240576)
#define OFF_NORMV  0u          // 32*4
#define OFF_EV     256u        // 32*30*4 = 3840
#define OFF_H16    4096u       // 32*512*2 = 32768

// ---------------------------------------------------------------------------
__device__ __forceinline__ float wave_reduce(float v) {
#pragma unroll
  for (int o = 16; o > 0; o >>= 1) v += __shfl_down(v, o, 32);
  return v;
}

__device__ __forceinline__ float block_reduce(float v, float* red, int tid, int nthreads) {
  v = wave_reduce(v);
  if ((tid & 31) == 0) red[tid >> 5] = v;
  __syncthreads();
  float r = 0.f;
  const int nw = nthreads >> 5;
  if (tid < 32) {
    r = (tid < nw) ? red[tid] : 0.f;
    r = wave_reduce(r);
    if (tid == 0) red[0] = r;
  }
  __syncthreads();
  r = red[0];
  __syncthreads();
  return r;
}

// Gate schedule of ansatz14 (per layer of 40 gates):
//   RY(0..9) ; CRX(9->0, 8->9, ..., 0->1) ; RY(0..9) ; CRX(9->8, 0->9, 1->0, ..., 8->7)
// Applied to a 1024-amplitude state held in LDS. blockDim must be 512.
__device__ __forceinline__ void apply_ansatz(float2* st, const float2* cs, int ngates, int tid) {
  for (int g = 0; g < ngates; ++g) {
    const int j40 = g % 40;
    const float c = cs[g].x, s = cs[g].y;
    if (j40 < 10 || (j40 >= 20 && j40 < 30)) {           // RY: 512 pairs
      const int w    = (j40 < 10) ? j40 : (j40 - 20);
      const int mask = 1 << (9 - w);
      const int lo   = mask - 1;
      const int i0   = ((tid & ~lo) << 1) | (tid & lo);
      const int i1   = i0 | mask;
      const float2 A = st[i0], B = st[i1];
      st[i0] = make_float2(c * A.x - s * B.x, c * A.y - s * B.y);
      st[i1] = make_float2(s * A.x + c * B.x, s * A.y + c * B.y);
    } else if (tid < 256) {                              // CRX: 256 pairs
      int ctrl, tgt;
      if (j40 < 20) { ctrl = 9 - (j40 - 10); tgt = (ctrl + 1) % 10; }
      else { const int jj = j40 - 30; ctrl = (jj == 0) ? 9 : jj - 1; tgt = (ctrl + 9) % 10; }
      const int mc = 1 << (9 - ctrl), mt = 1 << (9 - tgt);
      const int lo = ((mc < mt) ? mc : mt) - 1;
      const int hi = ((mc < mt) ? mt : mc) - 1;
      int i = ((tid & ~lo) << 1) | (tid & lo);
      i     = ((i   & ~hi) << 1) | (i   & hi);
      const int i0 = i | mc;            // ctrl=1, tgt=0
      const int i1 = i0 | mt;           // ctrl=1, tgt=1
      const float2 A = st[i0], B = st[i1];
      // na = c*a - i*s*b ; nb = -i*s*a + c*b
      st[i0] = make_float2(c * A.x + s * B.y, c * A.y - s * B.x);
      st[i1] = make_float2(c * B.x + s * A.y, c * B.y - s * A.x);
    }
    __syncthreads();
  }
}

// ---------------------------------------------------------------------------
// angles[m][p] = sum_e emb[x[m]][e] * w_ang[p][e] + b_ang[p]
// One wave per 16x16 output tile. grid = (64 mtiles, 5 ntiles), block = 32.
__global__ void k_angles(const int* __restrict__ x, const float* __restrict__ emb,
                         const float* __restrict__ w_ang, const float* __restrict__ b_ang,
                         float* __restrict__ angles) {
  const int lane = threadIdx.x;
  const int Ml   = lane & 15;
  const int half = lane >> 4;
  const int row  = blockIdx.x * 16 + Ml;       // A row (token position)
  const int p    = blockIdx.y * 16 + Ml;       // B column (param index)
  const float* __restrict__ erow = emb + (size_t)x[row] * EMBD;
  const float* __restrict__ wrow = w_ang + (size_t)p * EMBD;
  v8f acc = {};
  for (int kk = 0; kk < 16; ++kk) {
    v16h a, b;
    const float* p0 = erow + kk * 32 + half * 8;
#pragma unroll
    for (int i = 0; i < 8; ++i) { a[i] = (_Float16)p0[i]; a[8 + i] = (_Float16)p0[16 + i]; }
    const float* q0 = wrow + kk * 32 + half * 8;
#pragma unroll
    for (int i = 0; i < 8; ++i) { b[i] = (_Float16)q0[i]; b[8 + i] = (_Float16)q0[16 + i]; }
    acc = __builtin_amdgcn_wmma_f32_16x16x32_f16(false, a, false, b, (short)0, acc, false, false);
  }
  const float bias = b_ang[p];
#pragma unroll
  for (int r = 0; r < 8; ++r)
    angles[(size_t)(blockIdx.x * 16 + half * 8 + r) * NPAR + p] = acc[r] + bias;
}

// mono[b] = e0 ; acc[b] = coeffs[0] * e0
__global__ void k_init(float2* __restrict__ mono, float2* __restrict__ acc,
                       const float* __restrict__ coeffs) {
  const int b = blockIdx.x;
  const float c0 = coeffs[0];
  for (int i = threadIdx.x; i < DIMQ; i += 256) {
    const float re = (i == 0) ? 1.f : 0.f;
    mono[b * DIMQ + i] = make_float2(re, 0.f);
    acc[b * DIMQ + i]  = make_float2(c0 * re, 0.f);
  }
}

// One state (b,t) per block; LDS-resident 2-layer ansatz14 (80 gates).
__global__ void k_ansatz(const float2* __restrict__ mono, const float* __restrict__ angles,
                         float2* __restrict__ stout) {
  __shared__ float2 st[DIMQ];
  __shared__ float2 cs[80];
  const int s   = blockIdx.x;           // s = b*32 + t
  const int b   = s >> 5;
  const int tid = threadIdx.x;          // 0..511
  st[tid]       = mono[b * DIMQ + tid];
  st[tid + 512] = mono[b * DIMQ + tid + 512];
  if (tid < 80) {
    float sn, csn;
    __sincosf(0.5f * angles[(size_t)s * NPAR + tid], &sn, &csn);
    cs[tid] = make_float2(csn, sn);
  }
  __syncthreads();
  apply_ansatz(st, cs, 80, tid);
  stout[(size_t)s * DIMQ + tid]       = st[tid];
  stout[(size_t)s * DIMQ + tid + 512] = st[tid + 512];
}

// mono[b][i] = sum_t st[b][t][i]*lcu_n[t] ; acc += coeffs[k]*mono
__global__ void k_lcu(const float2* __restrict__ st, const float* __restrict__ lcu_re,
                      const float* __restrict__ lcu_im, const float* __restrict__ coeffs,
                      int k, float2* __restrict__ mono, float2* __restrict__ acc) {
  __shared__ float2 lc[NTOK];
  const int tid = threadIdx.x;
  const int i   = blockIdx.x * 256 + tid;
  const int b   = blockIdx.y;
  if (tid < NTOK) {
    float ssum = 0.f;
    for (int t = 0; t < NTOK; ++t)
      ssum += sqrtf(lcu_re[t] * lcu_re[t] + lcu_im[t] * lcu_im[t]);
    const float inv = 1.f / fmaxf(ssum, 1e-12f);
    lc[tid] = make_float2(lcu_re[tid] * inv, lcu_im[tid] * inv);
  }
  __syncthreads();
  float2 m = make_float2(0.f, 0.f);
  for (int t = 0; t < NTOK; ++t) {
    const float2 v = st[((size_t)(b * NTOK + t)) * DIMQ + i];
    const float2 w = lc[t];
    m.x += v.x * w.x - v.y * w.y;
    m.y += v.x * w.y + v.y * w.x;
  }
  mono[b * DIMQ + i] = m;
  const float ck = coeffs[k];
  float2 a = acc[b * DIMQ + i];
  acc[b * DIMQ + i] = make_float2(a.x + ck * m.x, a.y + ck * m.y);
}

// qsvt_state = acc / sum|coeffs| ; norm = ||.|| ; normed = qsvt/max(norm,eps)
__global__ void k_norm(const float2* __restrict__ acc, const float* __restrict__ coeffs,
                       float2* __restrict__ normed, float* __restrict__ normv) {
  __shared__ float red[8];
  const int b = blockIdx.x, tid = threadIdx.x;
  const float inv = 1.f / (fabsf(coeffs[0]) + fabsf(coeffs[1]) +
                           fabsf(coeffs[2]) + fabsf(coeffs[3]));
  float ss = 0.f;
  float2 v[4];
#pragma unroll
  for (int r = 0; r < 4; ++r) {
    const float2 a = acc[b * DIMQ + r * 256 + tid];
    v[r] = make_float2(a.x * inv, a.y * inv);
    ss += v[r].x * v[r].x + v[r].y * v[r].y;
  }
  const float nrm = sqrtf(block_reduce(ss, red, tid, 256));
  const float s = 1.f / fmaxf(nrm, 1e-12f);
#pragma unroll
  for (int r = 0; r < 4; ++r)
    normed[b * DIMQ + r * 256 + tid] = make_float2(v[r].x * s, v[r].y * s);
  if (tid == 0) normv[b] = nrm;
}

__global__ void k_mean(const float* __restrict__ normv, float* __restrict__ out) {
  float v = normv[threadIdx.x] * (1.f / 32.f);
  v = wave_reduce(v);
  if (threadIdx.x == 0) out[(size_t)NBATCH * VOCAB_] = v;
}

// qff ansatz (1 layer, 40 gates) + X/Y/Z expectation values -> ev[b][30]
__global__ void k_qff(const float2* __restrict__ normed, const float* __restrict__ qff,
                      float* __restrict__ ev) {
  __shared__ float2 st[DIMQ];
  __shared__ float2 cs[40];
  __shared__ float  red[16];
  const int b = blockIdx.x, tid = threadIdx.x;
  st[tid]       = normed[b * DIMQ + tid];
  st[tid + 512] = normed[b * DIMQ + tid + 512];
  if (tid < 40) {
    float sn, csn;
    __sincosf(0.5f * qff[tid], &sn, &csn);
    cs[tid] = make_float2(csn, sn);
  }
  __syncthreads();
  apply_ansatz(st, cs, 40, tid);
  for (int w = 0; w < 10; ++w) {
    const int mask = 1 << (9 - w), lo = mask - 1;
    const int i0 = ((tid & ~lo) << 1) | (tid & lo);
    const int i1 = i0 | mask;
    const float2 A = st[i0], B = st[i1];
    const float px = 2.f * (A.x * B.x + A.y * B.y);
    const float py = 2.f * (A.x * B.y - A.y * B.x);
    const float pz = (A.x * A.x + A.y * A.y) - (B.x * B.x + B.y * B.y);
    const float X = block_reduce(px, red, tid, 512);
    const float Y = block_reduce(py, red, tid, 512);
    const float Z = block_reduce(pz, red, tid, 512);
    if (tid == 0) { ev[b * 30 + w] = X; ev[b * 30 + 10 + w] = Y; ev[b * 30 + 20 + w] = Z; }
  }
}

// h[b][j] = relu(ev[b] . w1[j] + b1[j]) stored as f16 (WMMA A operand)
__global__ void k_h(const float* __restrict__ ev, const float* __restrict__ w1,
                    const float* __restrict__ b1, _Float16* __restrict__ h16) {
  __shared__ float e[30];
  const int b = blockIdx.x, j = threadIdx.x;
  if (j < 30) e[j] = ev[b * 30 + j];
  __syncthreads();
  float s = b1[j];
#pragma unroll
  for (int q = 0; q < 30; ++q) s += e[q] * w1[j * 30 + q];
  h16[b * EMBD + j] = (_Float16)fmaxf(s, 0.f);
}

// logits = h @ w2^T + b2. 4 waves/block, 1 ntile per wave, both M tiles per
// wave so each w2 tile (B operand) is loaded once. grid = 500, block = 128.
__global__ void k_logits(const _Float16* __restrict__ h16, const float* __restrict__ w2,
                         const float* __restrict__ b2, float* __restrict__ out) {
  const int lane  = threadIdx.x & 31;
  const int wave  = threadIdx.x >> 5;
  const int ntile = blockIdx.x * 4 + wave;
  const int Ml    = lane & 15;
  const int half  = lane >> 4;
  const int v     = ntile * 16 + Ml;                 // vocab column
  const float*    __restrict__ wrow = w2 + (size_t)v * EMBD;
  const _Float16* __restrict__ h0   = h16 + Ml * EMBD;
  const _Float16* __restrict__ h1   = h16 + (16 + Ml) * EMBD;
  v8f acc0 = {}, acc1 = {};
  for (int kk = 0; kk < 16; ++kk) {
    v16h a0, a1, bm;
    const int base = kk * 32 + half * 8;
#pragma unroll
    for (int i = 0; i < 8; ++i) {
      a0[i] = h0[base + i]; a0[8 + i] = h0[base + 16 + i];
      a1[i] = h1[base + i]; a1[8 + i] = h1[base + 16 + i];
    }
    const float* q0 = wrow + kk * 32 + half * 8;
#pragma unroll
    for (int i = 0; i < 8; ++i) { bm[i] = (_Float16)q0[i]; bm[8 + i] = (_Float16)q0[16 + i]; }
    acc0 = __builtin_amdgcn_wmma_f32_16x16x32_f16(false, a0, false, bm, (short)0, acc0, false, false);
    acc1 = __builtin_amdgcn_wmma_f32_16x16x32_f16(false, a1, false, bm, (short)0, acc1, false, false);
  }
  const float bias = b2[v];
#pragma unroll
  for (int r = 0; r < 8; ++r) {
    out[(size_t)(half * 8 + r) * VOCAB_ + v]      = acc0[r] + bias;
    out[(size_t)(16 + half * 8 + r) * VOCAB_ + v] = acc1[r] + bias;
  }
}

// ---------------------------------------------------------------------------
extern "C" void kernel_launch(void* const* d_in, const int* in_sizes, int n_in,
                              void* d_out, int out_size, void* d_ws, size_t ws_size,
                              hipStream_t stream) {
  const int*   x      = (const int*)d_in[0];
  const float* emb    = (const float*)d_in[1];
  const float* w_ang  = (const float*)d_in[2];
  const float* b_ang  = (const float*)d_in[3];
  const float* coeffs = (const float*)d_in[4];
  const float* lcu_re = (const float*)d_in[5];
  const float* lcu_im = (const float*)d_in[6];
  const float* qff    = (const float*)d_in[7];
  const float* w1     = (const float*)d_in[8];
  const float* b1     = (const float*)d_in[9];
  const float* w2     = (const float*)d_in[10];
  const float* b2     = (const float*)d_in[11];
  float* out = (float*)d_out;

  char* ws = (char*)d_ws;
  float*  angles = (float*)(ws + OFF_ANGLES);
  float2* mono   = (float2*)(ws + OFF_MONO);
  float2* acc    = (float2*)(ws + OFF_ACC);
  float2* st     = (float2*)(ws + OFF_ST);
  float2* normed = mono;                         // mono dead after loop
  float*  normv  = (float*)(ws + OFF_NORMV);     // overlays dead angles
  float*  ev     = (float*)(ws + OFF_EV);
  _Float16* h16  = (_Float16*)(ws + OFF_H16);

  k_angles<<<dim3(64, 5), 32, 0, stream>>>(x, emb, w_ang, b_ang, angles);
  k_init<<<NBATCH, 256, 0, stream>>>(mono, acc, coeffs);
  for (int k = 1; k <= 3; ++k) {
    k_ansatz<<<NBATCH * NTOK, 512, 0, stream>>>(mono, angles, st);
    k_lcu<<<dim3(4, NBATCH), 256, 0, stream>>>(st, lcu_re, lcu_im, coeffs, k, mono, acc);
  }
  k_norm<<<NBATCH, 256, 0, stream>>>(acc, coeffs, normed, normv);
  k_mean<<<1, 32, 0, stream>>>(normv, out);
  k_qff<<<NBATCH, 512, 0, stream>>>(normed, qff, ev);
  k_h<<<NBATCH, EMBD, 0, stream>>>(ev, w1, b1, h16);
  k_logits<<<VOCAB_ / 64, 128, 0, stream>>>(h16, w2, b2, out);
}
